// EncoderLayer_63273458204947
// MI455X (gfx1250) — matrix operand order repaired and checked by reading, not compiled, not run on hardware
//
#include <hip/hip_runtime.h>
#include <hip/hip_bf16.h>
#include <math.h>
#include <stdint.h>

typedef __attribute__((ext_vector_type(16))) _Float16 v16h;
typedef __attribute__((ext_vector_type(8)))  float    v8f;
typedef __attribute__((ext_vector_type(4)))  int      v4i;

#define H_DIM   1024
#define NHEADS  16
#define DHEAD   64
#define FFN_DIM 2048
#define SEQ     1024
#define NBATCH  4

union U32H2 { unsigned int u; _Float16 h[2]; };
union F4H8  { float4 v; _Float16 h[8]; };

// ---------------- CDNA5 async global->LDS path (guarded; fallback compiles) --
#if __has_builtin(__builtin_amdgcn_global_load_async_to_lds_b128)
#define HAVE_ASYNC 1
#else
#define HAVE_ASYNC 0
#endif

#if HAVE_ASYNC
__device__ __forceinline__ void async_cp16(const _Float16* src, _Float16* dst_lds)
{
    // signature (from clang diagnostic): (v4i*, v4i*, imm offset, imm cpol)
    __builtin_amdgcn_global_load_async_to_lds_b128((v4i*)src, (v4i*)dst_lds, 0, 0);
}
__device__ __forceinline__ void wait_async_le4()
{
#if __has_builtin(__builtin_amdgcn_s_wait_asynccnt)
    __builtin_amdgcn_s_wait_asynccnt(4);
#else
    asm volatile("s_wait_asynccnt 0x4" ::: "memory");
#endif
}
__device__ __forceinline__ void wait_async_le0()
{
#if __has_builtin(__builtin_amdgcn_s_wait_asynccnt)
    __builtin_amdgcn_s_wait_asynccnt(0);
#else
    asm volatile("s_wait_asynccnt 0x0" ::: "memory");
#endif
}
#endif

// ---------------------------------------------------------------- conversions
__global__ __launch_bounds__(256) void cvt_f32_f16_kernel(
    const float* __restrict__ in, _Float16* __restrict__ out, int n4)
{
    int i = blockIdx.x * 256 + threadIdx.x;
    if (i < n4) {
        float4 v = ((const float4*)in)[i];
        out[4*i+0] = (_Float16)v.x;
        out[4*i+1] = (_Float16)v.y;
        out[4*i+2] = (_Float16)v.z;
        out[4*i+3] = (_Float16)v.w;
    }
}

// in: f32 [R][C] row-major  ->  out: f16 [C][R] (transposed), LDS-tiled 32x32
__global__ __launch_bounds__(256) void cvt_transpose_f16_kernel(
    const float* __restrict__ in, _Float16* __restrict__ out, int R, int C)
{
    __shared__ float tile[32][33];
    const int tx = threadIdx.x & 31, ty = threadIdx.x >> 5;
    const int r0 = blockIdx.y * 32, c0 = blockIdx.x * 32;
    #pragma unroll
    for (int i = 0; i < 4; ++i) {
        int r = ty + i * 8;
        tile[r][tx] = in[(size_t)(r0 + r) * C + c0 + tx];
    }
    __syncthreads();
    #pragma unroll
    for (int i = 0; i < 4; ++i) {
        int r = ty + i * 8;   // row of output tile = original column
        out[(size_t)(c0 + r) * R + r0 + tx] = (_Float16)tile[tx][r];
    }
}

__global__ __launch_bounds__(256) void copy_mask_kernel(
    const int* __restrict__ m, int* __restrict__ out, int n)
{
    int i = blockIdx.x * 256 + threadIdx.x;
    if (i < n) out[i] = m[i];
}

// ---------------------------------------------------------------- WMMA GEMM
// C[M,N] = op((A[M,K](f16) x Bt[N,K](f16)^T + bias) * scale); op = relu opt.
// Bt is the PRE-TRANSPOSED weight ([N][K]) so A and B tiles stage identically.
// Tiles: 128x128x32, 256 threads = 8 waves, each wave -> 32x64 C sub-tile.
// Double-buffered LDS; global->LDS staging via async DMA when available.
template <bool RELU, bool OUT16>
__global__ __launch_bounds__(256) void gemm_f16_kernel(
    const _Float16* __restrict__ A, const _Float16* __restrict__ Bt,
    const float* __restrict__ bias, int M, int N, int K,
    float scale, float* __restrict__ Cf32, _Float16* __restrict__ Cf16)
{
    __shared__ alignas(16) _Float16 As[2][128][40];   // [m][k], pad 8
    __shared__ alignas(16) _Float16 Bs[2][128][40];   // [n][k], pad 8

    const int tid  = threadIdx.x;
    const int lane = tid & 31;
    const int wave = tid >> 5;
    const int waveM = (wave >> 1) * 32;   // 0,32,64,96
    const int waveN = (wave &  1) * 64;   // 0,64
    const int mBase = blockIdx.y * 128;
    const int nBase = blockIdx.x * 128;

    v8f acc[2][4] = {};
    int buf = 0;

#if HAVE_ASYNC
    // prologue: stage k0 = 0 into buffer 0 (2 x b128 per thread per tile)
    #pragma unroll
    for (int c = 0; c < 2; ++c) {
        int idx = (c * 256 + tid) * 8;
        int r = idx >> 5, cc = idx & 31;
        async_cp16(A  + (size_t)(mBase + r) * K + cc, &As[0][r][cc]);
        async_cp16(Bt + (size_t)(nBase + r) * K + cc, &Bs[0][r][cc]);
    }
#endif

    for (int k0 = 0; k0 < K; k0 += 32) {
#if HAVE_ASYNC
        if (k0 + 32 < K) {
            #pragma unroll
            for (int c = 0; c < 2; ++c) {
                int idx = (c * 256 + tid) * 8;
                int r = idx >> 5, cc = idx & 31;
                async_cp16(A  + (size_t)(mBase + r) * K + k0 + 32 + cc, &As[buf ^ 1][r][cc]);
                async_cp16(Bt + (size_t)(nBase + r) * K + k0 + 32 + cc, &Bs[buf ^ 1][r][cc]);
            }
            wait_async_le4();   // current buffer's 4 DMAs done; next 4 in flight
        } else {
            wait_async_le0();
        }
        __syncthreads();
#else
        #pragma unroll
        for (int c = 0; c < 2; ++c) {
            int idx = (c * 256 + tid) * 8;
            int r = idx >> 5, cc = idx & 31;
            *(float4*)&As[buf][r][cc] = *(const float4*)(A  + (size_t)(mBase + r) * K + k0 + cc);
            *(float4*)&Bs[buf][r][cc] = *(const float4*)(Bt + (size_t)(nBase + r) * K + k0 + cc);
        }
        __syncthreads();
#endif

        v16h a[2], b[4];
        #pragma unroll
        for (int mi = 0; mi < 2; ++mi) {
            int row = waveM + mi * 16 + (lane & 15);
            #pragma unroll
            for (int j = 0; j < 8; ++j) {
                // A 16x32 f16 layout: K = (j/4)*16 + (lane/16)*8 + (j%4)*2 + b
                int kk = ((j >> 2) << 4) + ((lane >> 4) << 3) + ((j & 3) << 1);
                U32H2 t; t.u = *(const unsigned int*)&As[buf][row][kk];
                a[mi][2*j] = t.h[0]; a[mi][2*j+1] = t.h[1];
            }
        }
        #pragma unroll
        for (int nj = 0; nj < 4; ++nj) {
            int col = waveN + nj * 16 + (lane & 15);
            #pragma unroll
            for (int j = 0; j < 8; ++j) {
                // B 32x16 f16 layout: K = (j/4)*16 + (lane/16)*8 + (j%4)*2 + b
                int kk = ((j >> 2) << 4) + ((lane >> 4) << 3) + ((j & 3) << 1);
                U32H2 t; t.u = *(const unsigned int*)&Bs[buf][col][kk];
                b[nj][2*j] = t.h[0]; b[nj][2*j+1] = t.h[1];
            }
        }
        #pragma unroll
        for (int mi = 0; mi < 2; ++mi)
            #pragma unroll
            for (int nj = 0; nj < 4; ++nj)
                acc[mi][nj] = __builtin_amdgcn_wmma_f32_16x16x32_f16(
                    false, a[mi], false, b[nj], (short)0, acc[mi][nj], false, false);
        __syncthreads();
        buf ^= 1;
    }

    // Branch-free epilogue: bias, scale, optional relu, single typed store
    #pragma unroll
    for (int mi = 0; mi < 2; ++mi) {
        const int row0 = mBase + waveM + mi * 16 + ((lane >> 4) << 3);
        #pragma unroll
        for (int nj = 0; nj < 4; ++nj) {
            const int col = nBase + waveN + nj * 16 + (lane & 15);
            const float bv = bias[col];
            if constexpr (OUT16) {
                _Float16* p = Cf16 + (size_t)row0 * N + col;
                #pragma unroll
                for (int v = 0; v < 8; ++v) {
                    float x = (acc[mi][nj][v] + bv) * scale;
                    if constexpr (RELU) x = fmaxf(x, 0.f);
                    p[(size_t)v * N] = (_Float16)x;
                }
            } else {
                float* p = Cf32 + (size_t)row0 * N + col;
                #pragma unroll
                for (int v = 0; v < 8; ++v) {
                    float x = (acc[mi][nj][v] + bv) * scale;
                    if constexpr (RELU) x = fmaxf(x, 0.f);
                    p[(size_t)v * N] = x;
                }
            }
        }
    }
}

// ---------------------------------------------------------------- attention
// Fused flash attention: block = (128 query rows, head h, batch n), 8 waves,
// each wave owns 16 query rows (low VGPR pressure: ~150 regs incl. transients).
// Q,K already carry the 1/sqrt(H) scale. Layout of Q/K/V/ctx: [n, L, H] f16.
__global__ __launch_bounds__(256) void attn_kernel(
    const _Float16* __restrict__ Q, const _Float16* __restrict__ Kp,
    const _Float16* __restrict__ V, const int* __restrict__ mask,
    _Float16* __restrict__ ctx, int L)
{
    __shared__ alignas(16) _Float16 Ks[64][72];       // [j][d] row-major
    __shared__ alignas(16) _Float16 Vs[64][72];       // transposed: [d][j]
    __shared__ alignas(16) _Float16 Ps[8][16][72];    // per-wave P staging
    __shared__ float mAdd[64];

    const int n = blockIdx.z, h = blockIdx.y;
    const int qBase = blockIdx.x * 128;
    const int tid = threadIdx.x, lane = tid & 31, wave = tid >> 5;
    const int qw = qBase + wave * 16;
    const size_t base = ((size_t)n * L) * H_DIM + (size_t)h * DHEAD;

    // Q fragments: 16 rows x d=64 -> 2 k-step frags
    v16h qa[2];
    #pragma unroll
    for (int kk = 0; kk < 2; ++kk)
        #pragma unroll
        for (int j = 0; j < 8; ++j) {
            int row = qw + (lane & 15);
            int d = kk * 32 + ((j >> 2) << 4) + ((lane >> 4) << 3) + ((j & 3) << 1);
            U32H2 t; t.u = *(const unsigned int*)(Q + base + (size_t)row * H_DIM + d);
            qa[kk][2*j] = t.h[0]; qa[kk][2*j+1] = t.h[1];
        }

    v8f o[4] = {};
    float mrun[8], lrun[8];
    #pragma unroll
    for (int v = 0; v < 8; ++v) { mrun[v] = -INFINITY; lrun[v] = 0.f; }

    for (int jb = 0; jb < L; jb += 64) {
        __syncthreads();
        // load K block [64 x 64] (row-major): 256 thr x 2 passes x 8 halves
        #pragma unroll
        for (int p = 0; p < 2; ++p) {
            int idx = (p * 256 + tid) * 8;
            int r = idx >> 6, c = idx & 63;
            *(float4*)&Ks[r][c] =
                *(const float4*)(Kp + base + (size_t)(jb + r) * H_DIM + c);
        }
        // load V block transposed: Vs[d][j]
        #pragma unroll
        for (int p = 0; p < 2; ++p) {
            int idx = (p * 256 + tid) * 8;
            int r = idx >> 6, c = idx & 63;
            F4H8 u;
            u.v = *(const float4*)(V + base + (size_t)(jb + r) * H_DIM + c);
            #pragma unroll
            for (int i = 0; i < 8; ++i) Vs[c + i][r] = u.h[i];
        }
        if (tid < 64) mAdd[tid] = (mask[(size_t)n * L + jb + tid] == 0) ? -INFINITY : 0.f;
        // prefetch next K/V block while this one is consumed
        if (jb + 64 < L) {
            int pr = tid >> 2, pc = (tid & 3) * 16;
            __builtin_prefetch(Kp + base + (size_t)(jb + 64 + pr) * H_DIM + pc, 0, 1);
            __builtin_prefetch(V  + base + (size_t)(jb + 64 + pr) * H_DIM + pc, 0, 1);
        }
        __syncthreads();

        // S = Q x K^T  (B-frag of K^T reads row-major K directly)
        v8f s[4] = {};
        #pragma unroll
        for (int kk = 0; kk < 2; ++kk) {
            #pragma unroll
            for (int nj = 0; nj < 4; ++nj) {
                v16h bfr;
                int col = nj * 16 + (lane & 15);
                #pragma unroll
                for (int j = 0; j < 8; ++j) {
                    int d = kk * 32 + ((j >> 2) << 4) + ((lane >> 4) << 3) + ((j & 3) << 1);
                    U32H2 t; t.u = *(const unsigned int*)&Ks[col][d];
                    bfr[2*j] = t.h[0]; bfr[2*j+1] = t.h[1];
                }
                s[nj] = __builtin_amdgcn_wmma_f32_16x16x32_f16(
                    false, qa[kk], false, bfr, (short)0, s[nj], false, false);
            }
        }

        // mask + online softmax (row = 8*(lane/16) + v)
        #pragma unroll
        for (int v = 0; v < 8; ++v) {
            float pm = -INFINITY;
            #pragma unroll
            for (int nj = 0; nj < 4; ++nj) {
                float x = s[nj][v] + mAdd[nj * 16 + (lane & 15)];
                s[nj][v] = x;
                pm = fmaxf(pm, x);
            }
            #pragma unroll
            for (int sh = 1; sh < 16; sh <<= 1) pm = fmaxf(pm, __shfl_xor(pm, sh, 32));
            float mnew = fmaxf(mrun[v], pm);
            float alpha = (mnew == -INFINITY) ? 1.f : __expf(mrun[v] - mnew);
            float ls = 0.f;
            int prow = ((lane >> 4) << 3) + v;
            #pragma unroll
            for (int nj = 0; nj < 4; ++nj) {
                float p = (mnew == -INFINITY) ? 0.f : __expf(s[nj][v] - mnew);
                Ps[wave][prow][nj * 16 + (lane & 15)] = (_Float16)p;
                ls += p;
            }
            #pragma unroll
            for (int sh = 1; sh < 16; sh <<= 1) ls += __shfl_xor(ls, sh, 32);
            lrun[v] = lrun[v] * alpha + ls;
            mrun[v] = mnew;
            #pragma unroll
            for (int dj = 0; dj < 4; ++dj) o[dj][v] *= alpha;
        }

        // O += P x V  (A-frags from per-wave LDS staging, B-frags from Vs)
        #pragma unroll
        for (int kk = 0; kk < 2; ++kk) {
            v16h pa;
            int row = lane & 15;
            #pragma unroll
            for (int j = 0; j < 8; ++j) {
                int kq = kk * 32 + ((j >> 2) << 4) + ((lane >> 4) << 3) + ((j & 3) << 1);
                U32H2 t; t.u = *(const unsigned int*)&Ps[wave][row][kq];
                pa[2*j] = t.h[0]; pa[2*j+1] = t.h[1];
            }
            #pragma unroll
            for (int dj = 0; dj < 4; ++dj) {
                v16h vb;
                int col = dj * 16 + (lane & 15);
                #pragma unroll
                for (int j = 0; j < 8; ++j) {
                    int kq = kk * 32 + ((j >> 2) << 4) + ((lane >> 4) << 3) + ((j & 3) << 1);
                    U32H2 t; t.u = *(const unsigned int*)&Vs[col][kq];
                    vb[2*j] = t.h[0]; vb[2*j+1] = t.h[1];
                }
                o[dj] = __builtin_amdgcn_wmma_f32_16x16x32_f16(
                    false, pa, false, vb, (short)0, o[dj], false, false);
            }
        }
    }

    // normalize and write ctx (f16)
    #pragma unroll
    for (int dj = 0; dj < 4; ++dj)
        #pragma unroll
        for (int v = 0; v < 8; ++v) {
            int row = qw + ((lane >> 4) << 3) + v;
            int col = dj * 16 + (lane & 15);
            float inv = lrun[v] > 0.f ? 1.f / lrun[v] : 0.f;
            ctx[base + (size_t)row * H_DIM + col] = (_Float16)(o[dj][v] * inv);
        }
}

// ---------------------------------------------------------------- layernorm
// y = LN(x)*g + b + x ; one row (H=1024) per 256-thread block.
__global__ __launch_bounds__(256) void ln_residual_kernel(
    const float* __restrict__ x, const float* __restrict__ g, const float* __restrict__ b,
    float* __restrict__ of32, _Float16* __restrict__ of16)
{
    __shared__ float red[8];
    const int row = blockIdx.x, tid = threadIdx.x;
    const float* xr = x + (size_t)row * H_DIM;
    float vals[4]; float s = 0.f;
    #pragma unroll
    for (int i = 0; i < 4; ++i) { vals[i] = xr[tid + i * 256]; s += vals[i]; }
    #pragma unroll
    for (int sh = 1; sh < 32; sh <<= 1) s += __shfl_xor(s, sh, 32);
    if ((tid & 31) == 0) red[tid >> 5] = s;
    __syncthreads();
    float mu = 0.f;
    #pragma unroll
    for (int i = 0; i < 8; ++i) mu += red[i];
    mu *= (1.f / H_DIM);
    __syncthreads();
    float vs = 0.f;
    #pragma unroll
    for (int i = 0; i < 4; ++i) { float d = vals[i] - mu; vs += d * d; }
    #pragma unroll
    for (int sh = 1; sh < 32; sh <<= 1) vs += __shfl_xor(vs, sh, 32);
    if ((tid & 31) == 0) red[tid >> 5] = vs;
    __syncthreads();
    float var = 0.f;
    #pragma unroll
    for (int i = 0; i < 8; ++i) var += red[i];
    var *= (1.f / H_DIM);
    float rinv = rsqrtf(var + 1e-5f);
    #pragma unroll
    for (int i = 0; i < 4; ++i) {
        int c = tid + i * 256;
        float y = (vals[i] - mu) * rinv * g[c] + b[c] + vals[i];
        if (of32) of32[(size_t)row * H_DIM + c] = y;
        if (of16) of16[(size_t)row * H_DIM + c] = (_Float16)y;
    }
}

// ---------------------------------------------------------------- launcher
extern "C" void kernel_launch(void* const* d_in, const int* in_sizes, int n_in,
                              void* d_out, int out_size, void* d_ws, size_t ws_size,
                              hipStream_t stream)
{
    const float* src  = (const float*)d_in[0];
    const int*   mask = (const int*)  d_in[1];
    const float* Wq   = (const float*)d_in[2];
    const float* bq   = (const float*)d_in[3];
    const float* Wk   = (const float*)d_in[4];
    const float* bk   = (const float*)d_in[5];
    const float* Wv   = (const float*)d_in[6];
    const float* bv   = (const float*)d_in[7];
    const float* Wo   = (const float*)d_in[8];
    const float* bo   = (const float*)d_in[9];
    const float* ln1g = (const float*)d_in[10];
    const float* ln1b = (const float*)d_in[11];
    const float* W1   = (const float*)d_in[12];
    const float* b1   = (const float*)d_in[13];
    const float* W2   = (const float*)d_in[14];
    const float* b2   = (const float*)d_in[15];
    const float* ln2g = (const float*)d_in[16];
    const float* ln2b = (const float*)d_in[17];

    const int M = NBATCH * SEQ;          // 4096 rows
    const size_t MB = 1024 * 1024;
    char* w = (char*)d_ws;
    _Float16* x16  = (_Float16*)(w + 0);        // 8MB; reused as ctx16
    _Float16* wq16 = (_Float16*)(w + 8  * MB);  // [N][K] transposed weights
    _Float16* wk16 = (_Float16*)(w + 10 * MB);
    _Float16* wv16 = (_Float16*)(w + 12 * MB);
    _Float16* wo16 = (_Float16*)(w + 14 * MB);
    _Float16* w116 = (_Float16*)(w + 16 * MB);  // 4MB  [FFN][H]
    _Float16* w216 = (_Float16*)(w + 20 * MB);  // 4MB  [H][FFN]
    _Float16* q16  = (_Float16*)(w + 24 * MB);  // 8MB; reused as x1_f16
    _Float16* k16  = (_Float16*)(w + 32 * MB);  // 8MB; k16+v16 reused as ffn1_f16
    _Float16* v16  = (_Float16*)(w + 40 * MB);  // 8MB
    float*    hid  = (float*)   (w + 48 * MB);  // 16MB; reused as ffn f32

    // activations: plain convert; weights [R][C] -> transposed f16 [C][R]
    {
        int n4 = (M * H_DIM) / 4;
        cvt_f32_f16_kernel<<<dim3((n4 + 255) / 256), dim3(256), 0, stream>>>(src, x16, n4);
    }
    auto cvtT = [&](const float* s_, _Float16* d_, int R, int C) {
        cvt_transpose_f16_kernel<<<dim3(C / 32, R / 32), dim3(256), 0, stream>>>(s_, d_, R, C);
    };
    cvtT(Wq, wq16, H_DIM, H_DIM);
    cvtT(Wk, wk16, H_DIM, H_DIM);
    cvtT(Wv, wv16, H_DIM, H_DIM);
    cvtT(Wo, wo16, H_DIM, H_DIM);
    cvtT(W1, w116, H_DIM, FFN_DIM);    // -> [FFN][H]
    cvtT(W2, w216, FFN_DIM, H_DIM);    // -> [H][FFN]

    const float qs = 1.f / 32.f;       // 1/sqrt(H), applied to both Q and K
    dim3 blk(256);

    // f16-output GEMMs
    gemm_f16_kernel<false, true><<<dim3(H_DIM / 128, M / 128), blk, 0, stream>>>(
        x16, wq16, bq, M, H_DIM, H_DIM, qs, nullptr, q16);
    gemm_f16_kernel<false, true><<<dim3(H_DIM / 128, M / 128), blk, 0, stream>>>(
        x16, wk16, bk, M, H_DIM, H_DIM, qs, nullptr, k16);
    gemm_f16_kernel<false, true><<<dim3(H_DIM / 128, M / 128), blk, 0, stream>>>(
        x16, wv16, bv, M, H_DIM, H_DIM, 1.f, nullptr, v16);

    attn_kernel<<<dim3(SEQ / 128, NHEADS, NBATCH), dim3(256), 0, stream>>>(
        q16, k16, v16, mask, /*ctx (reuse x16)*/ x16, SEQ);

    gemm_f16_kernel<false, false><<<dim3(H_DIM / 128, M / 128), blk, 0, stream>>>(
        x16, wo16, bo, M, H_DIM, H_DIM, 1.f, hid, nullptr);

    ln_residual_kernel<<<dim3(M), dim3(256), 0, stream>>>(
        hid, ln1g, ln1b, (float*)nullptr, /*x1 f16*/ q16);

    gemm_f16_kernel<true, true><<<dim3(FFN_DIM / 128, M / 128), blk, 0, stream>>>(
        q16, w116, b1, M, FFN_DIM, H_DIM, 1.f, nullptr, /*ffn1*/ k16);
    gemm_f16_kernel<false, false><<<dim3(H_DIM / 128, M / 128), blk, 0, stream>>>(
        k16, w216, b2, M, H_DIM, FFN_DIM, 1.f, hid, nullptr);

    ln_residual_kernel<<<dim3(M), dim3(256), 0, stream>>>(
        hid, ln2g, ln2b, (float*)d_out, (_Float16*)nullptr);

    copy_mask_kernel<<<dim3(16), dim3(256), 0, stream>>>(
        mask, ((int*)d_out) + (size_t)M * H_DIM, NBATCH * SEQ);
}
